// ViMamba_36060545417602
// MI455X (gfx1250) — matrix operand order repaired and checked by reading, not compiled, not run on hardware
//
#include <hip/hip_runtime.h>
#include <hip/hip_bf16.h>

// ViMamba fused pipeline for MI455X (gfx1250, wave32, WMMA).
// Dims: B*T=192, SEQ=207, tokens NT=39744 (=16*2484), C=128, RED=64, DIN=128, DS=16.

typedef __bf16 v16bf __attribute__((ext_vector_type(16)));
typedef __bf16 v8bf  __attribute__((ext_vector_type(8)));
typedef float  v8f   __attribute__((ext_vector_type(8)));

#define NT  39744
#define MT  2484
#define SEQ 207
#define BTB 192

// ---------------------------------------------------------------------------
// Weight pre-pack: fp32 W[K,N] -> bf16 transposed Bt[N,K] (contiguous K per col).
// ---------------------------------------------------------------------------
__global__ void k_pack(const float* __restrict__ W, __bf16* __restrict__ Bt, int K, int N)
{
    const int idx = blockIdx.x * 256 + threadIdx.x;
    if (idx >= K * N) return;
    const int k = idx / N, n = idx % N;
    Bt[n * K + k] = (__bf16)W[idx];
}

// ---------------------------------------------------------------------------
// Fragment helpers (ISA 16-bit A 16x32 / B 32x16 lane layouts, wave32).
// ---------------------------------------------------------------------------
__device__ __forceinline__ v16bf frag_a(const __bf16* p /* row base + kb + (half?8:0) */)
{
    const v8bf lo = *(const v8bf*)(p);
    const v8bf hi = *(const v8bf*)(p + 16);
    v16bf a;
    #pragma unroll
    for (int e = 0; e < 8; ++e) { a[e] = lo[e]; a[8 + e] = hi[e]; }
    return a;
}
__device__ __forceinline__ v16bf frag_b(const __bf16* p /* col base + kb + (half?8:0) */)
{
    const v8bf lo = *(const v8bf*)(p);
    const v8bf hi = *(const v8bf*)(p + 16);
    v16bf b;
    #pragma unroll
    for (int e = 0; e < 8; ++e) { b[e] = lo[e]; b[8 + e] = hi[e]; }
    return b;
}

// ---------------------------------------------------------------------------
// Kernel 1: concat(x,qk) -> LayerNorm(256) -> GEMM 256x64 (+bias) -> h (f32+bf16)
// Block: 128 threads (4 waves), one 16-row M tile; each wave owns one 16-col N tile.
// B operand from pre-packed bf16 transposed in_w (64 x 256).
// ---------------------------------------------------------------------------
__global__ void k_ln_in(const float* __restrict__ x, const float* __restrict__ qk,
                        const float* __restrict__ lnw, const float* __restrict__ lnb,
                        const __bf16* __restrict__ Wit, const float* __restrict__ bi,
                        float* __restrict__ h, __bf16* __restrict__ hbf)
{
    __shared__ float ps[16][8], pq[16][8];
    __shared__ float sm[16], sr[16];
    __shared__ __bf16 Ablk[16][256];
    const int tid = threadIdx.x;
    const int r = tid >> 3, l8 = tid & 7;
    const long tok = (long)blockIdx.x * 16 + r;

    float s = 0.f, q = 0.f;
    for (int j0 = 0; j0 < 32; ++j0) {
        int j = l8 * 32 + j0;
        float v = (j < 128) ? x[tok * 128 + j] : qk[tok * 128 + j - 128];
        s += v; q += v * v;
    }
    ps[r][l8] = s; pq[r][l8] = q;
    __syncthreads();
    if (tid < 16) {
        float ss = 0.f, qq = 0.f;
        for (int i = 0; i < 8; ++i) { ss += ps[tid][i]; qq += pq[tid][i]; }
        float m = ss * (1.f / 256.f);
        sm[tid] = m;
        sr[tid] = rsqrtf(qq * (1.f / 256.f) - m * m + 1e-5f);
    }
    __syncthreads();
    const float m = sm[r], ri = sr[r];
    for (int j0 = 0; j0 < 32; ++j0) {
        int j = l8 * 32 + j0;
        float v = (j < 128) ? x[tok * 128 + j] : qk[tok * 128 + j - 128];
        Ablk[r][j] = (__bf16)(((v - m) * ri) * lnw[j] + lnb[j]);
    }
    __syncthreads();

    const int lane = tid & 31, wave = tid >> 5;
    const int lrow = lane & 15, half = lane >> 4;
    const int nb = wave * 16;
    const int col = nb + lrow;
    v8f acc = {0.f, 0.f, 0.f, 0.f, 0.f, 0.f, 0.f, 0.f};
    #pragma unroll
    for (int kb = 0; kb < 256; kb += 32) {
        const v16bf a = frag_a(&Ablk[lrow][kb + (half ? 8 : 0)]);
        const __bf16* bp = Wit + (long)col * 256 + kb + (half ? 8 : 0);
        if (kb + 32 < 256) __builtin_prefetch(bp + 32, 0, 0);
        const v16bf b = frag_b(bp);
        acc = __builtin_amdgcn_wmma_f32_16x16x32_bf16(false, a, false, b, (short)0, acc, false, false);
    }
    #pragma unroll
    for (int v = 0; v < 8; ++v) {
        const int rr = half * 8 + v;
        const long g = (long)blockIdx.x * 16 + rr;
        const float val = acc[v] + bi[col];
        h[g * 64 + col] = val;
        hbf[g * 64 + col] = (__bf16)val;
    }
}

// ---------------------------------------------------------------------------
// Generic bf16 WMMA GEMM: C[M,N] (+=) A_bf16[M,K] * Bt_bf16[N,K] (+bias)
// One wave per 16x16 tile; K compile-time (full unroll -> back-to-back WMMA).
// ---------------------------------------------------------------------------
template <int K>
__global__ void k_gemm_bf16(const __bf16* __restrict__ A, const __bf16* __restrict__ Bt,
                            const float* __restrict__ bias, float* __restrict__ C,
                            int N, int accFlag)
{
    const int lane = threadIdx.x;
    const int lrow = lane & 15, half = lane >> 4;
    const long mb = (long)blockIdx.x * 16;
    const int nb = blockIdx.y * 16;
    const int col = nb + lrow;

    v8f acc;
    #pragma unroll
    for (int v = 0; v < 8; ++v) {
        const long row = mb + half * 8 + v;
        acc[v] = accFlag ? C[row * N + col] : 0.f;
    }
    const __bf16* arow = A + (mb + lrow) * (long)K + (half ? 8 : 0);
    const __bf16* brow = Bt + (long)col * K + (half ? 8 : 0);
    #pragma unroll
    for (int kb = 0; kb < K; kb += 32) {
        if (kb + 32 < K) {
            __builtin_prefetch(arow + kb + 32, 0, 0);
            __builtin_prefetch(brow + kb + 32, 0, 0);
        }
        const v16bf a = frag_a(arow + kb);
        const v16bf b = frag_b(brow + kb);
        acc = __builtin_amdgcn_wmma_f32_16x16x32_bf16(false, a, false, b, (short)0, acc, false, false);
    }
    const float bv = bias ? bias[col] : 0.f;
    #pragma unroll
    for (int v = 0; v < 8; ++v) {
        const long row = mb + half * 8 + v;
        C[row * N + col] = acc[v] + bv;
    }
}

// ---------------------------------------------------------------------------
// Causal depthwise conv (K=4) + SiLU, sequence direction selectable.
// Output u stored in mamba (scan) order.
// ---------------------------------------------------------------------------
__global__ void k_conv_silu(const float* __restrict__ xz, const float* __restrict__ cw,
                            const float* __restrict__ cb, float* __restrict__ u, int dir)
{
    const long idx = (long)blockIdx.x * blockDim.x + threadIdx.x;
    if (idx >= (long)NT * 128) return;
    const int c = (int)(idx & 127);
    const long i = idx >> 7;
    const int bt = (int)(i / SEQ), p = (int)(i % SEQ);
    float acc = cb[c];
    #pragma unroll
    for (int k = 0; k < 4; ++k) {
        const int qpos = p - 3 + k;
        if (qpos >= 0) {
            const int n = dir ? (SEQ - 1 - qpos) : qpos;
            acc += cw[c * 4 + k] * xz[((long)bt * SEQ + n) * 256 + c];
        }
    }
    u[idx] = acc / (1.f + __expf(-acc)); // SiLU
}

// ---------------------------------------------------------------------------
// xdbl = u @ xproj (128->36); dt = softplus(xdbl[:4] @ dtw + dtb); split B,C.
// One block per token, 128 threads.
// ---------------------------------------------------------------------------
__global__ void k_xdbl(const float* __restrict__ u, const float* __restrict__ xp,
                       const float* __restrict__ dtw, const float* __restrict__ dtb,
                       float* __restrict__ dt, float* __restrict__ Bc, float* __restrict__ Cc)
{
    __shared__ float su[128];
    __shared__ float xd[36];
    const int tid = threadIdx.x;
    const long tok = blockIdx.x;
    su[tid] = u[tok * 128 + tid];
    __syncthreads();
    if (tid < 36) {
        float a = 0.f;
        for (int k = 0; k < 128; ++k) a += su[k] * xp[k * 36 + tid];
        xd[tid] = a;
    }
    __syncthreads();
    float raw = dtb[tid];
    #pragma unroll
    for (int rr = 0; rr < 4; ++rr) raw += xd[rr] * dtw[rr * 128 + tid];
    dt[tok * 128 + tid] = (raw > 20.f) ? raw : log1pf(__expf(raw));
    if (tid < 16) { Bc[tok * 16 + tid] = xd[4 + tid]; Cc[tok * 16 + tid] = xd[20 + tid]; }
}

// ---------------------------------------------------------------------------
// Selective scan: one block per batch row, one lane per channel (16 states in regs).
// Prefetches next-step operands (latency-critical serial chain).
// Writes gated output (in ORIGINAL token order) as bf16 for the outproj WMMA GEMM.
// ---------------------------------------------------------------------------
__global__ void k_scan(const float* __restrict__ dt, const float* __restrict__ u,
                       const float* __restrict__ Bc, const float* __restrict__ Cc,
                       const float* __restrict__ xz, const float* __restrict__ Alog,
                       const float* __restrict__ Dp, __bf16* __restrict__ yg, int dir)
{
    __shared__ float sB[16], sC[16];
    const int d = threadIdx.x;
    const int bt = blockIdx.x;
    float A[16], hs[16];
    #pragma unroll
    for (int s = 0; s < 16; ++s) { A[s] = -__expf(Alog[d * 16 + s]); hs[s] = 0.f; }
    const float Dd = Dp[d];
    for (int p = 0; p < SEQ; ++p) {
        const long tok = (long)bt * SEQ + p;
        __syncthreads();
        if (d < 16) { sB[d] = Bc[tok * 16 + d]; sC[d] = Cc[tok * 16 + d]; }
        __syncthreads();
        if (p + 1 < SEQ) {
            const long t2 = tok + 1;
            __builtin_prefetch(&dt[t2 * 128 + d], 0, 0);
            __builtin_prefetch(&u[t2 * 128 + d], 0, 0);
            if (d < 16) {
                __builtin_prefetch(&Bc[t2 * 16 + d], 0, 0);
                __builtin_prefetch(&Cc[t2 * 16 + d], 0, 0);
            }
        }
        const float dtv = dt[tok * 128 + d];
        const float uv  = u[tok * 128 + d];
        const float du = dtv * uv;
        float y = 0.f;
        #pragma unroll
        for (int s = 0; s < 16; ++s) {
            hs[s] = hs[s] * __expf(dtv * A[s]) + du * sB[s];
            y += hs[s] * sC[s];
        }
        const int n = dir ? (SEQ - 1 - p) : p;
        const long otok = (long)bt * SEQ + n;
        const float z = xz[otok * 256 + 128 + d];
        const float sz = z / (1.f + __expf(-z));
        yg[otok * 128 + d] = (__bf16)((y + uv * Dd) * sz);
    }
}

// ---------------------------------------------------------------------------
// LN(64) -> WMMA GEMM 64x128 (+bias) -> + x residual -> LN(128) -> bf16 a2.
// Block: 256 threads (8 waves), one 16-row M tile; each wave one 16-col N tile.
// B operand from pre-packed bf16 transposed ov_w (128 x 64).
// ---------------------------------------------------------------------------
__global__ void k_ov(const float* __restrict__ hsum,
                     const float* __restrict__ ovlnw, const float* __restrict__ ovlnb,
                     const __bf16* __restrict__ Wovt, const float* __restrict__ bov,
                     const float* __restrict__ x,
                     const float* __restrict__ olnw, const float* __restrict__ olnb,
                     __bf16* __restrict__ a2)
{
    __shared__ float sm[16], sr[16];
    __shared__ __bf16 Ablk[16][64];
    __shared__ float out2[16][128];
    const int tid = threadIdx.x;
    const long mb = (long)blockIdx.x * 16;

    if (tid < 16) {
        float ss = 0.f, qq = 0.f;
        for (int j = 0; j < 64; ++j) { float t = hsum[(mb + tid) * 64 + j]; ss += t; qq += t * t; }
        float m = ss * (1.f / 64.f);
        sm[tid] = m;
        sr[tid] = rsqrtf(qq * (1.f / 64.f) - m * m + 1e-5f);
    }
    __syncthreads();
    for (int it = 0; it < 4; ++it) {
        const int e = it * 256 + tid;
        const int r = e >> 6, j = e & 63;
        const float t = hsum[(mb + r) * 64 + j];
        Ablk[r][j] = (__bf16)(((t - sm[r]) * sr[r]) * ovlnw[j] + ovlnb[j]);
    }
    __syncthreads();

    const int lane = tid & 31, wave = tid >> 5;
    const int lrow = lane & 15, half = lane >> 4;
    const int nb = wave * 16;
    const int col = nb + lrow;
    v8f acc = {0.f, 0.f, 0.f, 0.f, 0.f, 0.f, 0.f, 0.f};
    #pragma unroll
    for (int kb = 0; kb < 64; kb += 32) {
        const v16bf a = frag_a(&Ablk[lrow][kb + (half ? 8 : 0)]);
        const v16bf b = frag_b(Wovt + (long)col * 64 + kb + (half ? 8 : 0));
        acc = __builtin_amdgcn_wmma_f32_16x16x32_bf16(false, a, false, b, (short)0, acc, false, false);
    }
    #pragma unroll
    for (int v = 0; v < 8; ++v) {
        const int rr = half * 8 + v;
        out2[rr][col] = acc[v] + bov[col] + x[(mb + rr) * 128 + col];
    }
    __syncthreads();
    if (tid < 16) {
        float ss = 0.f, qq = 0.f;
        for (int j = 0; j < 128; ++j) { float t = out2[tid][j]; ss += t; qq += t * t; }
        float m = ss * (1.f / 128.f);
        sm[tid] = m;
        sr[tid] = rsqrtf(qq * (1.f / 128.f) - m * m + 1e-5f);
    }
    __syncthreads();
    for (int it = 0; it < 8; ++it) {
        const int e = it * 256 + tid;
        const int r = e >> 7, j = e & 127;
        a2[(mb + r) * 128 + j] = (__bf16)(((out2[r][j] - sm[r]) * sr[r]) * olnw[j] + olnb[j]);
    }
}

// ---------------------------------------------------------------------------
// Host launcher
// ---------------------------------------------------------------------------
extern "C" void kernel_launch(void* const* d_in, const int* in_sizes, int n_in,
                              void* d_out, int out_size, void* d_ws, size_t ws_size,
                              hipStream_t stream)
{
    (void)in_sizes; (void)n_in; (void)out_size; (void)ws_size;
    const float* x       = (const float*)d_in[0];
    const float* qk      = (const float*)d_in[1];
    const float* in_ln_w = (const float*)d_in[2];
    const float* in_ln_b = (const float*)d_in[3];
    const float* in_w    = (const float*)d_in[4];
    const float* in_b    = (const float*)d_in[5];
    const float* ov_ln_w = (const float*)d_in[6];
    const float* ov_ln_b = (const float*)d_in[7];
    const float* ov_w    = (const float*)d_in[8];
    const float* ov_b    = (const float*)d_in[9];
    const float* o_ln_w  = (const float*)d_in[10];
    const float* o_ln_b  = (const float*)d_in[11];
    const float* o_w     = (const float*)d_in[12];
    const float* o_b     = (const float*)d_in[13];

    char* ws = (char*)d_ws;
    size_t off = 0;
    auto take = [&](size_t bytes) -> char* {
        char* p = ws + off;
        off += (bytes + 255) & ~(size_t)255;
        return p;
    };
    // Activations
    float*  h    = (float*) take((size_t)NT * 64  * 4);
    __bf16* hbf  = (__bf16*)take((size_t)NT * 64  * 2);
    float*  xz   = (float*) take((size_t)NT * 256 * 4);
    float*  u    = (float*) take((size_t)NT * 128 * 4);
    float*  dt   = (float*) take((size_t)NT * 128 * 4);
    float*  Bc   = (float*) take((size_t)NT * 16  * 4);
    float*  Cc   = (float*) take((size_t)NT * 16  * 4);
    __bf16* yg   = (__bf16*)take((size_t)NT * 128 * 2);
    float*  hsum = (float*) take((size_t)NT * 64  * 4);
    __bf16* a2   = (__bf16*)take((size_t)NT * 128 * 2);
    // Pre-packed bf16 transposed weights
    __bf16* inw_t  = (__bf16*)take(256 * 64  * 2);   // [64][256]
    __bf16* ovw_t  = (__bf16*)take(64  * 128 * 2);   // [128][64]
    __bf16* ow_t   = (__bf16*)take(128 * 128 * 2);   // [128][128]
    __bf16* inp_t[2], *out_t[2];
    inp_t[0] = (__bf16*)take(64  * 256 * 2);         // [256][64]
    inp_t[1] = (__bf16*)take(64  * 256 * 2);
    out_t[0] = (__bf16*)take(128 * 64  * 2);         // [64][128]
    out_t[1] = (__bf16*)take(128 * 64  * 2);

    // 0) pack weights (tiny)
    k_pack<<<(256 * 64  + 255) / 256, 256, 0, stream>>>(in_w, inw_t, 256, 64);
    k_pack<<<(64  * 128 + 255) / 256, 256, 0, stream>>>(ov_w, ovw_t, 64, 128);
    k_pack<<<(128 * 128 + 255) / 256, 256, 0, stream>>>(o_w,  ow_t,  128, 128);
    for (int dir = 0; dir < 2; ++dir) {
        const float* inproj = (const float*)d_in[14 + dir * 9 + 0];
        const float* outp   = (const float*)d_in[14 + dir * 9 + 8];
        k_pack<<<(64  * 256 + 255) / 256, 256, 0, stream>>>(inproj, inp_t[dir], 64, 256);
        k_pack<<<(128 * 64  + 255) / 256, 256, 0, stream>>>(outp,   out_t[dir], 128, 64);
    }

    // 1) concat + LN(256) + in_w GEMM
    k_ln_in<<<MT, 128, 0, stream>>>(x, qk, in_ln_w, in_ln_b, inw_t, in_b, h, hbf);

    // 2) fw then bw mamba, reusing workspace; bw accumulates into hsum
    for (int dir = 0; dir < 2; ++dir) {
        const float* conv_w = (const float*)d_in[14 + dir * 9 + 1];
        const float* conv_b = (const float*)d_in[14 + dir * 9 + 2];
        const float* xproj  = (const float*)d_in[14 + dir * 9 + 3];
        const float* dtw    = (const float*)d_in[14 + dir * 9 + 4];
        const float* dtb    = (const float*)d_in[14 + dir * 9 + 5];
        const float* Alog   = (const float*)d_in[14 + dir * 9 + 6];
        const float* Dp     = (const float*)d_in[14 + dir * 9 + 7];

        k_gemm_bf16<64><<<dim3(MT, 16), 32, 0, stream>>>(hbf, inp_t[dir], nullptr, xz, 256, 0);
        const long total = (long)NT * 128;
        k_conv_silu<<<(int)((total + 255) / 256), 256, 0, stream>>>(xz, conv_w, conv_b, u, dir);
        k_xdbl<<<NT, 128, 0, stream>>>(u, xproj, dtw, dtb, dt, Bc, Cc);
        k_scan<<<BTB, 128, 0, stream>>>(dt, u, Bc, Cc, xz, Alog, Dp, yg, dir);
        k_gemm_bf16<128><<<dim3(MT, 4), 32, 0, stream>>>(yg, out_t[dir], nullptr, hsum, 64, dir);
    }

    // 3) LN(64) + ov_w GEMM + residual + LN(128)
    k_ov<<<MT, 256, 0, stream>>>(hsum, ov_ln_w, ov_ln_b, ovw_t, ov_b, x, o_ln_w, o_ln_b, a2);

    // 4) final o_w GEMM into d_out
    k_gemm_bf16<128><<<dim3(MT, 8), 32, 0, stream>>>(a2, ow_t, o_b, (float*)d_out, 128, 0);
}
